// ConvTrans_decoder_79173427135061
// MI455X (gfx1250) — compile-verified, matrix operand order checked by reading, not run on hardware
//
#include <hip/hip_runtime.h>

// ---------------- problem constants ----------------
constexpr int NB = 8192;     // batch
constexpr int NE = 20000;    // entities
constexpr int NH = 512;      // hidden
constexpr int NC = 50;       // conv channels
constexpr float EPSF = 1e-5f;
constexpr long BH = (long)NB * NH;        // 4194304
constexpr int KDIM1 = NC * NH;            // 25600

// ---------------- workspace layout (bytes) ----------------
constexpr size_t OFF_STATS0 = 0;                         // 6 f  (sum0..2, sq0..2)
constexpr size_t OFF_SC0    = 256;                       // 3 f
constexpr size_t OFF_SH0    = 512;                       // 3 f
constexpr size_t OFF_STATS1 = 1024;                      // 100 f (sum[50], sq[50])
constexpr size_t OFF_SC1    = 2048;                      // 50 f
constexpr size_t OFF_SH1    = 4096;                      // 50 f
constexpr size_t OFF_STATS2 = 8192;                      // 1024 f (sum[512], sq[512])
constexpr size_t OFF_SC2    = 16384;                     // 512 f
constexpr size_t OFF_SH2    = 24576;                     // 512 f
constexpr size_t OFF_ROWSUB = 32768;                     // 8192 f (per-row lse)
constexpr size_t OFF_HIDDEN = 65536;                     // NB*NH f32   = 16 MB
constexpr size_t OFF_HBF    = OFF_HIDDEN + (size_t)NB*NH*4;   // NB*NH bf16 = 8 MB
constexpr size_t OFF_Y      = OFF_HBF + (size_t)NB*NH*2;      // NB*NC*NH bf16 = 420 MB

// ---------------- bf16 helpers (storage = unsigned short) ----------------
__device__ __forceinline__ float bf2f(unsigned short h) {
    unsigned int u = ((unsigned int)h) << 16;
    float f; __builtin_memcpy(&f, &u, 4); return f;
}
__device__ __forceinline__ unsigned short f2bf(float f) {
    unsigned int u; __builtin_memcpy(&u, &f, 4);
    u = (u + 0x7FFFu + ((u >> 16) & 1u)) >> 16;   // round-to-nearest-even
    return (unsigned short)u;
}
__device__ __forceinline__ unsigned int pack2(float a, float b) {
    return (unsigned int)f2bf(a) | ((unsigned int)f2bf(b) << 16);
}

// ---------------- WMMA vector types ----------------
typedef __attribute__((ext_vector_type(16))) __bf16 v16bf;
typedef __attribute__((ext_vector_type(8)))  float  v8f;

union Frag16 { uint4 q[2]; v16bf v; };

// Build one 16x32(A) / 32x16(B) bf16 fragment from an 80B-padded LDS row.
// Per ISA layout: lanes 0-15 khalf=0 -> K = {0..7} U {16..23};
//                 lanes 16-31 khalf=1 -> K = {8..15} U {24..31}.
__device__ __forceinline__ v16bf load_frag(const unsigned short* row, int khalf) {
    Frag16 f;
    f.q[0] = *(const uint4*)(row + khalf * 8);        // 8 halves, 16B aligned
    f.q[1] = *(const uint4*)(row + 16 + khalf * 8);   // 8 halves, 16B aligned
    return f.v;
}

// ---------------- kernel 0: zero stat accumulators ----------------
__global__ void ct_zero_stats(float* ws_f) {
    for (int i = threadIdx.x; i < 3072; i += blockDim.x) ws_f[i] = 0.f;  // bytes [0,12288)
}

// ---------------- kernel 1: BN0 statistics (gathered 3 channels) ----------------
__global__ __launch_bounds__(256)
void ct_bn0_stats(const int* __restrict__ facts, const float* __restrict__ ent,
                  const float* __restrict__ rel, const float* __restrict__ erb,
                  float* __restrict__ stats0) {
    __shared__ float part[6];
    int t = threadIdx.x;
    if (t < 6) part[t] = 0.f;
    __syncthreads();
    float s0 = 0, s1 = 0, s2 = 0, q0 = 0, q1 = 0, q2 = 0;
    for (long i = (long)blockIdx.x * 256 + t; i < BH; i += (long)gridDim.x * 256) {
        int b = (int)(i >> 9), h = (int)(i & 511);
        int es = facts[b * 3 + 0];
        int er = facts[b * 3 + 1];
        float v0 = ent[(long)es * NH + h];
        float v1 = rel[(long)er * NH + h];
        float v2 = erb[(long)es * NH + h];
        s0 += v0; q0 += v0 * v0;
        s1 += v1; q1 += v1 * v1;
        s2 += v2; q2 += v2 * v2;
    }
    atomicAdd(&part[0], s0); atomicAdd(&part[1], s1); atomicAdd(&part[2], s2);
    atomicAdd(&part[3], q0); atomicAdd(&part[4], q1); atomicAdd(&part[5], q2);
    __syncthreads();
    if (t < 6) atomicAdd(&stats0[t], part[t]);
}

__global__ void ct_fin0(const float* __restrict__ stats0, const float* __restrict__ g,
                        const float* __restrict__ b, float* sc0, float* sh0) {
    int t = threadIdx.x;
    if (t >= 3) return;
    float cnt = (float)BH;
    float mean = stats0[t] / cnt;
    float var  = stats0[3 + t] / cnt - mean * mean;
    float s = g[t] * rsqrtf(var + EPSF);
    sc0[t] = s;
    sh0[t] = b[t] - mean * s;
}

// ---------------- kernel 2: gather + BN0 + conv1d + bias -> y(bf16), BN1 stats ----------------
__global__ __launch_bounds__(512)
void ct_conv(const int* __restrict__ facts, const float* __restrict__ ent,
             const float* __restrict__ rel, const float* __restrict__ erb,
             const float* __restrict__ conv_w, const float* __restrict__ conv_b,
             const float* __restrict__ sc0, const float* __restrict__ sh0,
             unsigned short* __restrict__ y, float* __restrict__ stats1) {
    __shared__ float xs[3][NH + 2];
    __shared__ float part[2 * NC];
    int b = blockIdx.x;
    int t = threadIdx.x;                 // t == h, 512 threads
    if (t < 2 * NC) part[t] = 0.f;
    int es = facts[b * 3 + 0];
    int er = facts[b * 3 + 1];
    xs[0][t + 1] = ent[(long)es * NH + t] * sc0[0] + sh0[0];
    xs[1][t + 1] = rel[(long)er * NH + t] * sc0[1] + sh0[1];
    xs[2][t + 1] = erb[(long)es * NH + t] * sc0[2] + sh0[2];
    if (t == 0) {
        #pragma unroll
        for (int i = 0; i < 3; ++i) { xs[i][0] = 0.f; xs[i][NH + 1] = 0.f; }
    }
    __syncthreads();
    float x0m = xs[0][t], x0c = xs[0][t + 1], x0p = xs[0][t + 2];
    float x1m = xs[1][t], x1c = xs[1][t + 1], x1p = xs[1][t + 2];
    float x2m = xs[2][t], x2c = xs[2][t + 1], x2p = xs[2][t + 2];
    long ybase = (long)b * KDIM1 + t;
    for (int c = 0; c < NC; ++c) {
        const float* w = conv_w + c * 9;
        float o = conv_b[c]
                + w[0] * x0m + w[1] * x0c + w[2] * x0p
                + w[3] * x1m + w[4] * x1c + w[5] * x1p
                + w[6] * x2m + w[7] * x2c + w[8] * x2p;
        y[ybase + (long)c * NH] = f2bf(o);
        atomicAdd(&part[c], o);
        atomicAdd(&part[NC + c], o * o);
    }
    __syncthreads();
    if (t < 2 * NC) atomicAdd(&stats1[t], part[t]);
}

__global__ void ct_fin1(const float* __restrict__ stats1, const float* __restrict__ g,
                        const float* __restrict__ b, float* sc1, float* sh1) {
    int t = threadIdx.x;
    if (t >= NC) return;
    float cnt = (float)BH;
    float mean = stats1[t] / cnt;
    float var  = stats1[NC + t] / cnt - mean * mean;
    float s = g[t] * rsqrtf(var + EPSF);
    sc1[t] = s;
    sh1[t] = b[t] - mean * s;
}

// ---------------- WMMA GEMM: Out[M,N] = op(A)[M,K] * B[N,K]^T (+bias) ----------------
// A: bf16 (ushort), optionally channel-affine (BN1) + ReLU at stage time.
// B: f32 rows, converted to bf16 at stage time.  Block tile 128x128x32, 8 waves.
template<bool A_AFFINE, bool N_GUARD, bool ADD_BIAS>
__global__ __launch_bounds__(256)
void ct_gemm(const unsigned short* __restrict__ Abf, const float* __restrict__ Bf,
             float* __restrict__ Out, const float* __restrict__ bias,
             const float* __restrict__ sc, const float* __restrict__ sh,
             int N, int Kdim) {
    constexpr int BM = 128, BN = 128, BK = 32;
    __shared__ __align__(16) unsigned short As[BM][40];   // 80B rows (padded)
    __shared__ __align__(16) unsigned short Bs[BN][40];

    const int nblk = (N + BN - 1) / BN;
    const int m0 = (blockIdx.x / nblk) * BM;
    const int n0 = (blockIdx.x % nblk) * BN;
    const int t = threadIdx.x;
    const int wave = t >> 5, lane = t & 31;
    const int wm = wave >> 1, wn = wave & 1;      // waves: 4 (M) x 2 (N)
    const int lm = lane & 15, khalf = lane >> 4;

    v8f acc[2][4];
    #pragma unroll
    for (int i = 0; i < 2; ++i)
        #pragma unroll
        for (int j = 0; j < 4; ++j)
            #pragma unroll
            for (int r = 0; r < 8; ++r) acc[i][j][r] = 0.f;

    for (int k0 = 0; k0 < Kdim; k0 += BK) {
        // ---- stage A tile (128 x 32 bf16) ----
        #pragma unroll
        for (int it = 0; it < 2; ++it) {
            int chunk = t + it * 256;
            int row = chunk >> 2, cc = chunk & 3;
            long gk = (long)k0 + cc * 8;
            const unsigned short* srcA = Abf + (long)(m0 + row) * Kdim + gk;
            uint4 d = *(const uint4*)srcA;
            if (k0 + BK < Kdim) __builtin_prefetch(srcA + BK, 0, 3);
            if (A_AFFINE) {
                int ch = (int)(gk >> 9);              // column/512 -> conv channel
                float s = sc[ch], o = sh[ch];
                unsigned int w[4] = {d.x, d.y, d.z, d.w};
                #pragma unroll
                for (int j = 0; j < 4; ++j) {
                    float lo = fmaxf(bf2f((unsigned short)(w[j] & 0xFFFFu)) * s + o, 0.f);
                    float hi = fmaxf(bf2f((unsigned short)(w[j] >> 16)) * s + o, 0.f);
                    w[j] = (unsigned int)f2bf(lo) | ((unsigned int)f2bf(hi) << 16);
                }
                d = make_uint4(w[0], w[1], w[2], w[3]);
            }
            *(uint4*)(&As[row][cc * 8]) = d;
        }
        // ---- stage B tile (128 n-rows x 32 k f32 -> bf16) ----
        #pragma unroll
        for (int it = 0; it < 2; ++it) {
            int chunk = t + it * 256;
            int row = chunk >> 2, cc = chunk & 3;
            int gn = n0 + row;
            long gk = (long)k0 + cc * 8;
            uint4 d;
            if (!N_GUARD || gn < N) {
                const float4* srcB = (const float4*)(Bf + (long)gn * Kdim + gk);
                float4 f0 = srcB[0], f1 = srcB[1];
                if (k0 + BK < Kdim) __builtin_prefetch((const float*)srcB + BK, 0, 3);
                d = make_uint4(pack2(f0.x, f0.y), pack2(f0.z, f0.w),
                               pack2(f1.x, f1.y), pack2(f1.z, f1.w));
            } else {
                d = make_uint4(0u, 0u, 0u, 0u);
            }
            *(uint4*)(&Bs[row][cc * 8]) = d;
        }
        __syncthreads();

        v16bf af[2], bfv[4];
        #pragma unroll
        for (int mi = 0; mi < 2; ++mi)
            af[mi] = load_frag(&As[wm * 32 + mi * 16 + lm][0], khalf);
        #pragma unroll
        for (int ni = 0; ni < 4; ++ni)
            bfv[ni] = load_frag(&Bs[wn * 64 + ni * 16 + lm][0], khalf);

        #pragma unroll
        for (int mi = 0; mi < 2; ++mi)
            #pragma unroll
            for (int ni = 0; ni < 4; ++ni)
                acc[mi][ni] = __builtin_amdgcn_wmma_f32_16x16x32_bf16(
                    false, af[mi], false, bfv[ni], (short)0, acc[mi][ni], false, false);
        __syncthreads();
    }

    // ---- epilogue: C/D layout  M = r + (lane/16)*8, N = lane%16 ----
    #pragma unroll
    for (int mi = 0; mi < 2; ++mi) {
        #pragma unroll
        for (int ni = 0; ni < 4; ++ni) {
            int n = n0 + wn * 64 + ni * 16 + lm;
            if (N_GUARD && n >= N) continue;
            float bb = ADD_BIAS ? bias[n] : 0.f;
            #pragma unroll
            for (int r = 0; r < 8; ++r) {
                int m = m0 + wm * 32 + mi * 16 + r + khalf * 8;
                Out[(long)m * N + n] = acc[mi][ni][r] + bb;
            }
        }
    }
}

// ---------------- BN2 stats / finalize / apply ----------------
__global__ __launch_bounds__(512)
void ct_bn2_stats(const float* __restrict__ hidden, float* __restrict__ stats2) {
    int t = threadIdx.x;                       // column 0..511
    long r0 = (long)blockIdx.x * 128;
    float s = 0.f, q = 0.f;
    for (int i = 0; i < 128; ++i) {
        float v = hidden[(r0 + i) * NH + t];
        s += v; q += v * v;
    }
    atomicAdd(&stats2[t], s);
    atomicAdd(&stats2[NH + t], q);
}

__global__ void ct_fin2(const float* __restrict__ stats2, const float* __restrict__ g,
                        const float* __restrict__ b, float* sc2, float* sh2) {
    int t = threadIdx.x;                       // 512 threads
    float mean = stats2[t] / (float)NB;
    float var  = stats2[NH + t] / (float)NB - mean * mean;
    float s = g[t] * rsqrtf(var + EPSF);
    sc2[t] = s;
    sh2[t] = b[t] - mean * s;
}

__global__ __launch_bounds__(256)
void ct_apply_bn2(const float* __restrict__ hidden, const float* __restrict__ sc2,
                  const float* __restrict__ sh2, unsigned short* __restrict__ hbf) {
    long i = (long)blockIdx.x * 256 + threadIdx.x;
    int col = (int)(i & 511);
    float v = fmaxf(hidden[i] * sc2[col] + sh2[col], 0.f);
    hbf[i] = f2bf(v);
}

// ---------------- row-wise streaming logsumexp, then in-place log_softmax ----------------
__global__ __launch_bounds__(256)
void ct_row_lse(const float* __restrict__ Out, float* __restrict__ rowsub) {
    __shared__ float sm[256], ss[256];
    int t = threadIdx.x;
    const float* row = Out + (long)blockIdx.x * NE;
    float m = -__builtin_inff(), s = 0.f;
    for (int i = t; i < NE; i += 256) {
        float v = row[i];
        if (v > m) { s = s * __expf(m - v) + 1.f; m = v; }
        else       { s += __expf(v - m); }
    }
    sm[t] = m; ss[t] = s;
    __syncthreads();
    for (int off = 128; off > 0; off >>= 1) {
        if (t < off) {
            float m2 = sm[t + off], s2 = ss[t + off];
            float mo = fmaxf(sm[t], m2);
            ss[t] = ss[t] * __expf(sm[t] - mo) + s2 * __expf(m2 - mo);
            sm[t] = mo;
        }
        __syncthreads();
    }
    if (t == 0) rowsub[blockIdx.x] = sm[0] + __logf(ss[0]);
}

__global__ __launch_bounds__(256)
void ct_logsoftmax(float* __restrict__ Out, const float* __restrict__ rowsub) {
    long i4 = (long)blockIdx.x * 256 + threadIdx.x;      // float4 index; NE % 4 == 0
    int row = (int)((i4 * 4) / NE);
    float sub = rowsub[row];
    float4* p = (float4*)Out + i4;
    float4 v = *p;
    v.x -= sub; v.y -= sub; v.z -= sub; v.w -= sub;
    *p = v;
}

// ---------------- launcher ----------------
extern "C" void kernel_launch(void* const* d_in, const int* in_sizes, int n_in,
                              void* d_out, int out_size, void* d_ws, size_t ws_size,
                              hipStream_t stream) {
    const int*   facts  = (const int*)d_in[0];
    const float* ent    = (const float*)d_in[1];
    const float* rel    = (const float*)d_in[2];
    const float* erb    = (const float*)d_in[3];
    const float* conv_w = (const float*)d_in[4];
    const float* conv_b = (const float*)d_in[5];
    const float* fc_w   = (const float*)d_in[6];
    const float* fc_b   = (const float*)d_in[7];
    const float* bn0_g  = (const float*)d_in[8];
    const float* bn0_b  = (const float*)d_in[9];
    const float* bn1_g  = (const float*)d_in[10];
    const float* bn1_b  = (const float*)d_in[11];
    const float* bn2_g  = (const float*)d_in[12];
    const float* bn2_b  = (const float*)d_in[13];
    float* out = (float*)d_out;

    char* ws = (char*)d_ws;
    float* stats0 = (float*)(ws + OFF_STATS0);
    float* sc0    = (float*)(ws + OFF_SC0);
    float* sh0    = (float*)(ws + OFF_SH0);
    float* stats1 = (float*)(ws + OFF_STATS1);
    float* sc1    = (float*)(ws + OFF_SC1);
    float* sh1    = (float*)(ws + OFF_SH1);
    float* stats2 = (float*)(ws + OFF_STATS2);
    float* sc2    = (float*)(ws + OFF_SC2);
    float* sh2    = (float*)(ws + OFF_SH2);
    float* rowsub = (float*)(ws + OFF_ROWSUB);
    float* hidden = (float*)(ws + OFF_HIDDEN);
    unsigned short* hbf = (unsigned short*)(ws + OFF_HBF);
    unsigned short* y   = (unsigned short*)(ws + OFF_Y);

    // 0) zero stat accumulators (ws is not re-poisoned between replays)
    ct_zero_stats<<<1, 256, 0, stream>>>((float*)ws);
    // 1) BN0 stats + finalize
    ct_bn0_stats<<<2048, 256, 0, stream>>>(facts, ent, rel, erb, stats0);
    ct_fin0<<<1, 32, 0, stream>>>(stats0, bn0_g, bn0_b, sc0, sh0);
    // 2) gather + BN0 + conv + bias -> y (bf16), BN1 stats + finalize
    ct_conv<<<NB, NH, 0, stream>>>(facts, ent, rel, erb, conv_w, conv_b, sc0, sh0, y, stats1);
    ct_fin1<<<1, 64, 0, stream>>>(stats1, bn1_g, bn1_b, sc1, sh1);
    // 3) GEMM1: hidden = relu(bn1(y)) @ fc_w^T + fc_b   (M=8192,N=512,K=25600)
    {
        int nblk = NH / 128;                       // 4
        int grid = (NB / 128) * nblk;              // 256
        ct_gemm<true, false, true><<<grid, 256, 0, stream>>>(
            y, fc_w, hidden, fc_b, sc1, sh1, NH, KDIM1);
    }
    // 4) BN2 stats + finalize + apply (-> bf16 activations)
    ct_bn2_stats<<<NB / 128, NH, 0, stream>>>(hidden, stats2);
    ct_fin2<<<1, NH, 0, stream>>>(stats2, bn2_g, bn2_b, sc2, sh2);
    ct_apply_bn2<<<(int)(BH / 256), 256, 0, stream>>>(hidden, sc2, sh2, hbf);
    // 5) GEMM2: scores = hbf @ ent_emb^T   (M=8192,N=20000,K=512) -> d_out
    {
        int nblk = (NE + 127) / 128;               // 157
        int grid = (NB / 128) * nblk;              // 10048
        ct_gemm<false, true, false><<<grid, 256, 0, stream>>>(
            hbf, ent, out, nullptr, nullptr, nullptr, NE, NH);
    }
    // 6) log_softmax in place
    ct_row_lse<<<NB, 256, 0, stream>>>(out, rowsub);
    {
        long n4 = (long)NB * NE / 4;               // 40,960,000
        ct_logsoftmax<<<(int)(n4 / 256), 256, 0, stream>>>(out, rowsub);
    }
}
